// SiameseGCN_83597243449447
// MI455X (gfx1250) — compile-verified, matrix operand order checked by reading, not run on hardware
//
#include <hip/hip_runtime.h>
#include <hip/hip_bf16.h>
#include <math.h>

typedef __attribute__((ext_vector_type(16))) _Float16 v16h;
typedef __attribute__((ext_vector_type(8)))  _Float16 v8h;
typedef __attribute__((ext_vector_type(8)))  float    v8f;

#define IN_FEATS 128
#define H2 128
#define N_CLASSES 64
#define N_GRAPHS 128
#define LDA 136   // padded LDS row stride in halves: 272B => bank stride 4, conflict-free b128 reads

// ---------------------------------------------------------------- utilities
__global__ void zero_kernel(float* __restrict__ p, size_t n) {
    size_t i = (size_t)blockIdx.x * blockDim.x + threadIdx.x;
    size_t stride = (size_t)gridDim.x * blockDim.x;
    for (; i < n; i += stride) p[i] = 0.0f;
}

// W[k][n] -> Wht[n][k] in f16 (one-time, 16K elements)
__global__ void convert_w_kernel(const float* __restrict__ W, _Float16* __restrict__ Wht) {
    int i = blockIdx.x * blockDim.x + threadIdx.x;
    if (i < IN_FEATS * H2) {
        int k = i >> 7, n = i & 127;
        Wht[n * IN_FEATS + k] = (_Float16)W[i];
    }
}

// ---------------------------------------------------------------- degrees
__global__ void degree_kernel(const int* __restrict__ src, const int* __restrict__ dst,
                              float* __restrict__ dego, float* __restrict__ degi, int n_edges) {
    int e = blockIdx.x * blockDim.x + threadIdx.x;
    if (e >= n_edges) return;
    atomicAdd(&dego[src[e]], 1.0f);
    atomicAdd(&degi[dst[e]], 1.0f);
}

// ---------------------------------------------------------------- projection: h = (feat * deg_out^-1/2) @ W
// Block: 256 threads = 8 waves; 32-row strip per block; wave w -> column tile w,
// two 16-row tiles per wave sharing one B fragment per K step.
__global__ __launch_bounds__(256) void project_kernel(
    const float* __restrict__ feat, const float* __restrict__ dego,
    const _Float16* __restrict__ Wht, float* __restrict__ h, int n_nodes) {
    __shared__ __attribute__((aligned(16))) _Float16 lA[32 * LDA];        // ~8.5 KB
    __shared__ __attribute__((aligned(16))) _Float16 lWt[IN_FEATS * LDA]; // ~34 KB (transposed W)

    int tid = threadIdx.x;
    int rowBase = blockIdx.x * 32;
    if (rowBase >= n_nodes) return;  // block-uniform

    // stage transposed W: 128 rows x 16 v8h chunks
    {
        const v8h* s8 = (const v8h*)Wht;
        v8h* d8 = (v8h*)lWt;  // row stride = 17 v8h
        for (int i = tid; i < 128 * 16; i += 256) {
            int r = i >> 4, c = i & 15;
            d8[r * 17 + c] = s8[i];
        }
    }
    // stage scaled A strip: 32 rows x 16 chunks of 8 cols
    for (int idx = tid; idx < 32 * 16; idx += 256) {
        int r = idx >> 4;
        int c0 = (idx & 15) * 8;
        float d = dego[rowBase + r];
        float s = rsqrtf(d < 1.0f ? 1.0f : d);
        const float* fp = feat + (size_t)(rowBase + r) * IN_FEATS + c0;
        float4 f0 = *(const float4*)(fp);
        float4 f1 = *(const float4*)(fp + 4);
        v8h pk;
        pk[0] = (_Float16)(f0.x * s); pk[1] = (_Float16)(f0.y * s);
        pk[2] = (_Float16)(f0.z * s); pk[3] = (_Float16)(f0.w * s);
        pk[4] = (_Float16)(f1.x * s); pk[5] = (_Float16)(f1.y * s);
        pk[6] = (_Float16)(f1.z * s); pk[7] = (_Float16)(f1.w * s);
        *(v8h*)(&lA[r * LDA + c0]) = pk;
    }
    __syncthreads();

    int wave = tid >> 5;
    int lane = tid & 31;
    int colBase = wave * 16;
    int m    = lane & 15;            // A row / B col within tile
    int kgrp = (lane >> 4) * 8;      // lane-group K offset (CDNA5 16-bit A layout)

    v8f acc0 = {}, acc1 = {};
    for (int kt = 0; kt < 4; ++kt) {
        int k0 = kt * 32 + kgrp;     // frag elems 0..7 ; elems 8..15 at k0+16
        const _Float16* bp = &lWt[(colBase + m) * LDA + k0];
        v16h b = __builtin_shufflevector(*(const v8h*)bp, *(const v8h*)(bp + 16),
                                         0,1,2,3,4,5,6,7,8,9,10,11,12,13,14,15);
        const _Float16* a0p = &lA[m * LDA + k0];
        v16h a0 = __builtin_shufflevector(*(const v8h*)a0p, *(const v8h*)(a0p + 16),
                                          0,1,2,3,4,5,6,7,8,9,10,11,12,13,14,15);
        const _Float16* a1p = &lA[(16 + m) * LDA + k0];
        v16h a1 = __builtin_shufflevector(*(const v8h*)a1p, *(const v8h*)(a1p + 16),
                                          0,1,2,3,4,5,6,7,8,9,10,11,12,13,14,15);
        acc0 = __builtin_amdgcn_wmma_f32_16x16x32_f16(false, a0, false, b, (short)0, acc0, false, false);
        acc1 = __builtin_amdgcn_wmma_f32_16x16x32_f16(false, a1, false, b, (short)0, acc1, false, false);
    }

    // C/D layout: VGPR r -> row = r (+8 for lanes 16-31), col = lane&15
    int rowOff = (lane < 16) ? 0 : 8;
#pragma unroll
    for (int r = 0; r < 8; ++r) {
        h[(size_t)(rowBase + rowOff + r) * H2 + colBase + m]      = acc0[r];
        h[(size_t)(rowBase + 16 + rowOff + r) * H2 + colBase + m] = acc1[r];
    }
}

// ---------------------------------------------------------------- edge scatter-add: agg[dst] += h[src]
// Grid-stride, one wave per edge per iteration; lane moves 4 floats.
// Prefetch next iteration's gathered row (gfx1250 global_prefetch_b8).
__global__ __launch_bounds__(256) void edge_agg_kernel(
    const int* __restrict__ src, const int* __restrict__ dst,
    const float* __restrict__ h, float* __restrict__ agg, int n_edges) {
    int waveId = blockIdx.x * (blockDim.x >> 5) + (threadIdx.x >> 5);
    int lane = threadIdx.x & 31;
    int stride = gridDim.x * (blockDim.x >> 5);
    for (int e = waveId; e < n_edges; e += stride) {
        int sn = src[e], dn = dst[e];
        int en = e + stride;
        if (en < n_edges) {
            __builtin_prefetch(&h[(size_t)src[en] * H2 + lane * 4], 0, 1);
        }
        float4 v = *(const float4*)(h + (size_t)sn * H2 + lane * 4);
        float* ap = agg + (size_t)dn * H2 + lane * 4;
        atomicAdd(&ap[0], v.x);
        atomicAdd(&ap[1], v.y);
        atomicAdd(&ap[2], v.z);
        atomicAdd(&ap[3], v.w);
    }
}

// ---------------------------------------------------------------- per-node: scale + bias, L2 normalize, sigmoid, graph accumulate
__global__ __launch_bounds__(256) void node_post_kernel(
    const float* __restrict__ agg, const float* __restrict__ degi,
    const float* __restrict__ bias, const int* __restrict__ gid,
    float* __restrict__ hg, float* __restrict__ cnt, int n_nodes) {
    int node = blockIdx.x * (blockDim.x >> 5) + (threadIdx.x >> 5);
    if (node >= n_nodes) return;
    int lane = threadIdx.x & 31;
    float d = degi[node];
    float s = rsqrtf(d < 1.0f ? 1.0f : d);
    float4 v = *(const float4*)(agg + (size_t)node * H2 + lane * 4);
    float4 bb = *(const float4*)(bias + lane * 4);
    v.x = v.x * s + bb.x; v.y = v.y * s + bb.y;
    v.z = v.z * s + bb.z; v.w = v.w * s + bb.w;
    float ss = v.x * v.x + v.y * v.y + v.z * v.z + v.w * v.w;
#pragma unroll
    for (int off = 16; off > 0; off >>= 1) ss += __shfl_xor(ss, off, 32);
    float inv = 1.0f / fmaxf(sqrtf(ss), 1e-12f);
    // relu(sigmoid(x)) == sigmoid(x)
    v.x = 1.0f / (1.0f + expf(-v.x * inv));
    v.y = 1.0f / (1.0f + expf(-v.y * inv));
    v.z = 1.0f / (1.0f + expf(-v.z * inv));
    v.w = 1.0f / (1.0f + expf(-v.w * inv));
    int g = gid[node];
    float* hgp = hg + (size_t)g * H2 + lane * 4;
    atomicAdd(&hgp[0], v.x);
    atomicAdd(&hgp[1], v.y);
    atomicAdd(&hgp[2], v.z);
    atomicAdd(&hgp[3], v.w);
    if (lane == 0) atomicAdd(&cnt[g], 1.0f);
}

// ---------------------------------------------------------------- classifier + pairwise distance
__global__ __launch_bounds__(64) void classify_kernel(
    const float* __restrict__ hg1, const float* __restrict__ cnt1,
    const float* __restrict__ hg2, const float* __restrict__ cnt2,
    const float* __restrict__ Wc, const float* __restrict__ bc,
    float* __restrict__ out) {
    __shared__ float sh[N_CLASSES];
    int g = blockIdx.x;
    int c = threadIdx.x;
    float ic1 = 1.0f / fmaxf(cnt1[g], 1.0f);
    float ic2 = 1.0f / fmaxf(cnt2[g], 1.0f);
    float d1 = bc[c], d2 = bc[c];
    for (int k = 0; k < H2; ++k) {
        float w = Wc[k * N_CLASSES + c];
        d1 += hg1[(size_t)g * H2 + k] * ic1 * w;
        d2 += hg2[(size_t)g * H2 + k] * ic2 * w;
    }
    float diff = d1 - d2 + 1e-6f;
    sh[c] = diff * diff;
    __syncthreads();
    for (int s = 32; s > 0; s >>= 1) {
        if (c < s) sh[c] += sh[c + s];
        __syncthreads();
    }
    if (c == 0) out[g] = sqrtf(sh[0]);
}

// ---------------------------------------------------------------- host
extern "C" void kernel_launch(void* const* d_in, const int* in_sizes, int n_in,
                              void* d_out, int out_size, void* d_ws, size_t ws_size,
                              hipStream_t stream) {
    const float* feat1 = (const float*)d_in[0];
    const float* feat2 = (const float*)d_in[1];
    const int*   src1  = (const int*)d_in[2];
    const int*   dst1  = (const int*)d_in[3];
    const int*   gid1  = (const int*)d_in[4];
    const int*   src2  = (const int*)d_in[5];
    const int*   dst2  = (const int*)d_in[6];
    const int*   gid2  = (const int*)d_in[7];
    const float* W     = (const float*)d_in[8];
    const float* bvec  = (const float*)d_in[9];
    const float* Wc    = (const float*)d_in[10];
    const float* bc    = (const float*)d_in[11];
    float* out = (float*)d_out;

    int n_nodes = in_sizes[0] / IN_FEATS;
    int n_edges = in_sizes[2];

    char* ws = (char*)d_ws;
    size_t off = 0;
    auto wsalloc = [&](size_t bytes) -> void* {
        void* p = ws + off;
        off = (off + bytes + 255) & ~(size_t)255;
        return p;
    };
    _Float16* Wht = (_Float16*)wsalloc((size_t)IN_FEATS * H2 * sizeof(_Float16));
    float* h      = (float*)wsalloc((size_t)n_nodes * H2 * sizeof(float));
    float* agg    = (float*)wsalloc((size_t)n_nodes * H2 * sizeof(float));
    float* dego   = (float*)wsalloc((size_t)n_nodes * sizeof(float));
    float* degi   = (float*)wsalloc((size_t)n_nodes * sizeof(float));
    float* hg1    = (float*)wsalloc((size_t)N_GRAPHS * H2 * sizeof(float));
    float* hg2    = (float*)wsalloc((size_t)N_GRAPHS * H2 * sizeof(float));
    float* cnt1   = (float*)wsalloc((size_t)N_GRAPHS * sizeof(float));
    float* cnt2   = (float*)wsalloc((size_t)N_GRAPHS * sizeof(float));

    convert_w_kernel<<<(IN_FEATS * H2 + 255) / 256, 256, 0, stream>>>(W, Wht);

    const float* feats[2] = {feat1, feat2};
    const int*   srcs[2]  = {src1, src2};
    const int*   dsts[2]  = {dst1, dst2};
    const int*   gids[2]  = {gid1, gid2};
    float*       hgs[2]   = {hg1, hg2};
    float*       cnts[2]  = {cnt1, cnt2};

    for (int br = 0; br < 2; ++br) {
        size_t aggN = (size_t)n_nodes * H2;
        zero_kernel<<<4096, 256, 0, stream>>>(agg, aggN);
        zero_kernel<<<512, 256, 0, stream>>>(dego, (size_t)n_nodes);
        zero_kernel<<<512, 256, 0, stream>>>(degi, (size_t)n_nodes);
        zero_kernel<<<64, 256, 0, stream>>>(hgs[br], (size_t)N_GRAPHS * H2);
        zero_kernel<<<1, 128, 0, stream>>>(cnts[br], (size_t)N_GRAPHS);

        degree_kernel<<<(n_edges + 255) / 256, 256, 0, stream>>>(
            srcs[br], dsts[br], dego, degi, n_edges);

        project_kernel<<<(n_nodes + 31) / 32, 256, 0, stream>>>(
            feats[br], dego, Wht, h, n_nodes);

        edge_agg_kernel<<<12500, 256, 0, stream>>>(
            srcs[br], dsts[br], h, agg, n_edges);

        node_post_kernel<<<(n_nodes + 7) / 8, 256, 0, stream>>>(
            agg, degi, bvec, gids[br], hgs[br], cnts[br], n_nodes);
    }

    classify_kernel<<<N_GRAPHS, N_CLASSES, 0, stream>>>(
        hg1, cnt1, hg2, cnt2, Wc, bc, out);
}
